// SimpleCNN_89678917141049
// MI455X (gfx1250) — matrix operand order repaired and checked by reading, not compiled, not run on hardware
//
#include <hip/hip_runtime.h>

typedef __attribute__((ext_vector_type(16))) __bf16 v16bf;
typedef __attribute__((ext_vector_type(2)))  __bf16 v2bf;
typedef __attribute__((ext_vector_type(8)))  float  v8f;

#define BATCH      8192
#define LSEQ       40
#define CIN        300
#define COUT       64
#define TB         4                    // batches per workgroup
#define KPAD       928                  // 900 padded to 29*32
#define KCHUNKS    29
#define SLAB_ELEMS 12640                // 42*300 zero-framed + tail pad
#define SLAB_BYTES (SLAB_ELEMS*2)       // 25280
#define WT_BYTES   (COUT*KPAD*2)        // 118784
#define OFF_WT     0
#define OFF_X      WT_BYTES
#define OFF_POOL   (OFF_X + TB*SLAB_BYTES)   // 219904
#define POOL_BYTES (TB*1280*4)               // 20480
#define SMEM_BYTES (OFF_POOL + POOL_BYTES)   // 240384  (< 320KB/WGP)

union AFrag { v16bf v; uint2 q[4]; };
union BFrag { v16bf v; uint4 q[2]; };

__device__ __forceinline__ unsigned pack2bf(float a, float b) {
    v2bf p; p[0] = (__bf16)a; p[1] = (__bf16)b;
    return __builtin_bit_cast(unsigned, p);
}

extern __shared__ __align__(16) char smem[];

__global__ __launch_bounds__(256)
void conv_pool_fc_kernel(const float* __restrict__ x,
                         const float* __restrict__ conv_w,
                         const float* __restrict__ conv_b,
                         const float* __restrict__ fc_w,
                         const float* __restrict__ fc_b,
                         float* __restrict__ out)
{
    const int tid = threadIdx.x;
    const int wg  = blockIdx.x;

    // ---- Stage 0: zero the pad frames of each x slab (head row + tail) ----
    for (int z = tid; z < TB * 320; z += 256) {
        int bb = z / 320, r = z - bb * 320;
        unsigned byteoff = (r < 150) ? (unsigned)(r * 4)
                                     : (unsigned)(24600 + (r - 150) * 4);
        *(unsigned*)(smem + OFF_X + bb * SLAB_BYTES + byteoff) = 0u;
    }

    // ---- Stage 1: conv weights -> LDS, transposed Wt[o][k*300+i], bf16, K padded ----
    for (int pp = tid; pp < (COUT * KPAD) / 2; pp += 256) {
        int n  = pp / 464;
        int kk = (pp - n * 464) * 2;
        float a = 0.f, b = 0.f;
        if (kk < 900) {
            int k0 = kk / 300, i0 = kk - k0 * 300;
            a = conv_w[n * 900 + i0 * 3 + k0];
        }
        if (kk + 1 < 900) {
            int k1 = (kk + 1) / 300, i1 = (kk + 1) - k1 * 300;
            b = conv_w[n * 900 + i1 * 3 + k1];
        }
        *(unsigned*)(smem + OFF_WT + (n * KPAD + kk) * 2) = pack2bf(a, b);
    }

    // ---- Stage 2: x -> LDS slabs (fp32 -> bf16), interior rows 1..40 ----
    const float* xg = x + (size_t)wg * TB * (LSEQ * CIN);
    for (int q = tid; q < TB * (LSEQ * CIN / 2); q += 256) {
        int bb = q / 6000;
        int e2 = (q - bb * 6000) * 2;
        float a = xg[bb * 12000 + e2];
        float b = xg[bb * 12000 + e2 + 1];
        *(unsigned*)(smem + OFF_X + bb * SLAB_BYTES + 600 + e2 * 2) = pack2bf(a, b);
    }

    __syncthreads();

    // ---- Stage 3: implicit-im2col GEMM via v_wmma_f32_16x16x32_bf16 ----
    const int wave = tid >> 5, lane = tid & 31;
    const int nt  = wave & 3;      // N-tile: channels nt*16 .. nt*16+15
    const int wv  = wave >> 2;     // M-tile group: tiles {wv, wv+2, ..., wv+8}
    const int ml  = lane & 15;
    const int khv = lane >> 4;     // K-half select per ISA A/B layouts

    // B fragment base: lane n = lane%16 reads Wt[nt*16+n][kc*32 + 8*khv .. +7] and [kc*32 + 16 + 8*khv .. +7]
    const char* wtb = smem + OFF_WT + (unsigned)(nt * 16 + ml) * (KPAD * 2) + khv * 16;

    // A fragment bases: lane m = lane%16 reads im2col row (contiguous slab span)
    unsigned abase[5];
    #pragma unroll
    for (int t = 0; t < 5; ++t) {
        int mt = wv + 2 * t;
        int r  = mt * 16 + ml;         // row = local (batch,l) index, 0..159
        int bb = r / 40;
        int l  = r - bb * 40;
        abase[t] = (unsigned)(OFF_X + bb * SLAB_BYTES + l * 600 + khv * 16);
    }

    v8f acc[5] = {};
    for (int kc = 0; kc < KCHUNKS; ++kc) {
        BFrag bf;
        bf.q[0] = *(const uint4*)(wtb + kc * 64);
        bf.q[1] = *(const uint4*)(wtb + kc * 64 + 32);
        #pragma unroll
        for (int t = 0; t < 5; ++t) {
            AFrag af;
            const char* ab = smem + abase[t] + kc * 64;
            af.q[0] = *(const uint2*)(ab);
            af.q[1] = *(const uint2*)(ab + 8);
            af.q[2] = *(const uint2*)(ab + 32);
            af.q[3] = *(const uint2*)(ab + 40);
            acc[t] = __builtin_amdgcn_wmma_f32_16x16x32_bf16(
                false, af.v, false, bf.v, (short)0, acc[t], false, false);
        }
    }

    // ---- Stage 4: bias + ReLU + maxpool(2) -> pooled LDS [bb][ch*20+p] ----
    const int  ch   = nt * 16 + ml;          // D-matrix N index == lane%16
    const float bias = conv_b[ch];
    #pragma unroll
    for (int t = 0; t < 5; ++t) {
        int mt = wv + 2 * t;
        #pragma unroll
        for (int j = 0; j < 4; ++j) {        // pool pairs = VGPR pairs (2j, 2j+1)
            float y0 = fmaxf(acc[t][2 * j]     + bias, 0.f);
            float y1 = fmaxf(acc[t][2 * j + 1] + bias, 0.f);
            float pv = fmaxf(y0, y1);
            int grow = mt * 16 + 2 * j + 8 * khv;   // even row of the pair
            int bb   = grow / 40;
            int l    = grow - bb * 40;
            int p    = l >> 1;
            *(float*)(smem + OFF_POOL + bb * 5120 + (ch * 20 + p) * 4) = pv;
        }
    }

    __syncthreads();

    // ---- Stage 5: fused FC: out[b][j] = pooled[b] . fc_w[j] + fc_b[j] ----
    const int bbf = wave >> 1, jj = wave & 1;
    const float* prow = (const float*)(smem + OFF_POOL + bbf * 5120);
    const float* wrow = fc_w + jj * 1280;
    float s = 0.f;
    for (int f = lane; f < 1280; f += 32)
        s += prow[f] * wrow[f];
    #pragma unroll
    for (int o = 16; o > 0; o >>= 1)
        s += __shfl_xor(s, o, 32);
    if (lane == 0)
        out[(size_t)(wg * TB + bbf) * 2 + jj] = s + fc_b[jj];
}

extern "C" void kernel_launch(void* const* d_in, const int* in_sizes, int n_in,
                              void* d_out, int out_size, void* d_ws, size_t ws_size,
                              hipStream_t stream) {
    (void)in_sizes; (void)n_in; (void)out_size; (void)d_ws; (void)ws_size;
    const float* x  = (const float*)d_in[0];
    const float* cw = (const float*)d_in[1];
    const float* cb = (const float*)d_in[2];
    const float* fw = (const float*)d_in[3];
    const float* fb = (const float*)d_in[4];
    float* out = (float*)d_out;

    hipFuncSetAttribute(reinterpret_cast<const void*>(conv_pool_fc_kernel),
                        hipFuncAttributeMaxDynamicSharedMemorySize, SMEM_BYTES);

    conv_pool_fc_kernel<<<dim3(BATCH / TB), dim3(256), SMEM_BYTES, stream>>>(
        x, cw, cb, fw, fb, out);
}
